// CaMoE_System_45457933861040
// MI455X (gfx1250) — matrix operand order repaired and checked by reading, not compiled, not run on hardware
//
#include <hip/hip_runtime.h>
#include <cstdint>
#include <cstddef>

// ---------------------------------------------------------------------------
// CaMoE forward for MI455X (gfx1250, wave32).
//  * All GEMMs: bf16 x bf16 -> f32 on v_wmma_f32_16x16x32_bf16.
//  * Weights converted once per launch to bf16 TRANSPOSED ([Npad][K], rows
//    zero-padded to a multiple of 64): every GEMM tile (A and B) is a
//    contiguous row-major 2-D strided tile.
//  * Tile staging via Tensor Data Mover (tensor_load_to_lds) with DOUBLE-
//    BUFFERED LDS: stage s+1 DMA overlaps stage s WMMA (wait tensorcnt<=2).
//    Manual vector-load path is the fallback when the builtin is absent.
//  * Sequential decay scan + top-2 routing stay scalar (tiny FLOP share).
// ---------------------------------------------------------------------------

typedef unsigned short u16;
typedef __attribute__((ext_vector_type(16))) __bf16        v16bf;
typedef __attribute__((ext_vector_type(8)))  float         v8f;
typedef __attribute__((ext_vector_type(4)))  unsigned int  v4u;

#define TILE_M 128
#define TILE_N 64
#define TILE_K 32

#if defined(__gfx1250__) && __has_builtin(__builtin_amdgcn_tensor_load_to_lds)
#define HAVE_TDM 1
#else
#define HAVE_TDM 0
#endif

namespace {
constexpr int B_  = 2, T_ = 1024, C_ = 768, V_ = 50257, L_ = 2;
constexpr int ER_ = 6, ET_ = 2, E_ = 8;
constexpr int H_  = 4 * C_;                    // 3072
constexpr int H2_ = 2 * C_;                    // 1536
constexpr int BT_ = B_ * T_;                   // 2048
constexpr int VP_ = (V_ + 63) & ~63;           // 50304: head N padded to 64
constexpr float EPS_ = 1e-5f;
}

__device__ __forceinline__ u16 f2bf(float f) {
  unsigned int u = __float_as_uint(f);
  u += 0x7FFFu + ((u >> 16) & 1u);             // round-to-nearest-even
  return (u16)(u >> 16);
}

// ---------------- conversion kernels ---------------------------------------

// plain f32 -> bf16 (activations)
__global__ void k_f2bf(const float* __restrict__ in, u16* __restrict__ out, size_t n) {
  size_t i = (size_t)blockIdx.x * blockDim.x + threadIdx.x;
  size_t stride = (size_t)gridDim.x * blockDim.x;
  for (; i < n; i += stride) out[i] = f2bf(in[i]);
}

// batched weight transpose+convert: in [batch][K][N] f32 -> out [batch][Npad][K]
// bf16, rows n >= N zero-filled.
__global__ void k_wT_bf16(const float* __restrict__ in, u16* __restrict__ out,
                          int batch, int K, int N, int Npad) {
  size_t per = (size_t)Npad * K;
  size_t total = (size_t)batch * per;
  size_t i = (size_t)blockIdx.x * blockDim.x + threadIdx.x;
  size_t stride = (size_t)gridDim.x * blockDim.x;
  for (; i < total; i += stride) {
    int b = (int)(i / per);
    size_t r = i % per;
    int n = (int)(r / K);
    int k = (int)(r % K);
    out[i] = (n < N) ? f2bf(in[(size_t)b * K * N + (size_t)k * N + n]) : (u16)0;
  }
}

// ---------------- elementwise kernels --------------------------------------

__global__ void k_embed(const int* __restrict__ idx, const float* __restrict__ emb,
                        float* __restrict__ x, int BT, int C) {
  size_t i = (size_t)blockIdx.x * blockDim.x + threadIdx.x;
  if (i >= (size_t)BT * C) return;
  int bt = (int)(i / C);
  int c  = (int)(i % C);
  x[i] = emb[(size_t)idx[bt] * C + c];
}

__global__ void k_vmix(float* __restrict__ v, float* __restrict__ vfirst,
                       int layer, size_t n) {
  size_t i = (size_t)blockIdx.x * blockDim.x + threadIdx.x;
  if (i >= n) return;
  if (layer == 0) vfirst[i] = v[i];
  else            v[i] = 0.5f * (v[i] + vfirst[i]);
}

__global__ void k_rs_bf(const float* __restrict__ r, const float* __restrict__ st,
                        u16* __restrict__ out, size_t n) {
  size_t i = (size_t)blockIdx.x * blockDim.x + threadIdx.x;
  if (i >= n) return;
  float sg = 1.f / (1.f + __expf(-r[i]));
  out[i] = f2bf(sg * st[i]);
}

__global__ void k_relu2_bf(const float* __restrict__ in, u16* __restrict__ out, size_t n) {
  size_t i = (size_t)blockIdx.x * blockDim.x + threadIdx.x;
  if (i >= n) return;
  float t = in[i];
  t = t > 0.f ? t : 0.f;
  out[i] = f2bf(t * t);
}

__global__ void k_silu_mul_bf(const float* __restrict__ a, const float* __restrict__ p,
                              u16* __restrict__ out, size_t n) {
  size_t i = (size_t)blockIdx.x * blockDim.x + threadIdx.x;
  if (i >= n) return;
  float av = a[i];
  float s  = av / (1.f + __expf(-av));
  out[i] = f2bf(s * p[i]);
}

__global__ void k_tanh_bf(const float* __restrict__ in, u16* __restrict__ out, size_t n) {
  size_t i = (size_t)blockIdx.x * blockDim.x + threadIdx.x;
  if (i >= n) return;
  out[i] = f2bf(tanhf(in[i]));
}

// ---------------- LayerNorm (one block of 256 per row) ---------------------

__global__ __launch_bounds__(256)
void k_layernorm(const float* __restrict__ in, const float* __restrict__ s,
                 const float* __restrict__ b, float* __restrict__ outf,
                 u16* __restrict__ outbf, int C) {
  int row = blockIdx.x;
  const float* xr = in + (size_t)row * C;
  __shared__ float red[256];
  float sum = 0.f, sq = 0.f;
  for (int c = threadIdx.x; c < C; c += 256) { float v = xr[c]; sum += v; sq += v * v; }
  red[threadIdx.x] = sum; __syncthreads();
  for (int o = 128; o > 0; o >>= 1) {
    if ((int)threadIdx.x < o) red[threadIdx.x] += red[threadIdx.x + o];
    __syncthreads();
  }
  float mean = red[0] / C;
  __syncthreads();
  red[threadIdx.x] = sq; __syncthreads();
  for (int o = 128; o > 0; o >>= 1) {
    if ((int)threadIdx.x < o) red[threadIdx.x] += red[threadIdx.x + o];
    __syncthreads();
  }
  float var = red[0] / C - mean * mean;
  float inv = rsqrtf(var + EPS_);
  for (int c = threadIdx.x; c < C; c += 256) {
    float y = (xr[c] - mean) * inv * s[c] + b[c];
    if (outf)  outf[(size_t)row * C + c]  = y;
    if (outbf) outbf[(size_t)row * C + c] = f2bf(y);
  }
}

// ---------------- decayed kv scan: s_t = d*s_{t-1} + k_t*v_t ---------------

__global__ void k_scan(const float* __restrict__ kin, const float* __restrict__ vin,
                       const float* __restrict__ decay, float* __restrict__ states,
                       int Bn, int Tn, int Cn) {
  int i = blockIdx.x * blockDim.x + threadIdx.x;
  if (i >= Bn * Cn) return;
  int b = i / Cn, c = i % Cn;
  float d = 1.f / (1.f + __expf(-decay[c]));
  size_t base = (size_t)b * Tn * Cn + c;
  float s = 0.f;
  for (int t = 0; t < Tn; ++t) {
    size_t o = base + (size_t)t * Cn;
    s = d * s + kin[o] * vin[o];
    states[o] = s;
  }
}

// ---------------- market routing: conf/affinity -> top2 -> gate ------------

__global__ void k_route(const float* __restrict__ h, const float* __restrict__ confw,
                        const float* __restrict__ confb, const float* __restrict__ cWa,
                        const float* __restrict__ cap, float* __restrict__ gate,
                        int BT, int C) {
  int t = blockIdx.x * blockDim.x + threadIdx.x;
  if (t >= BT) return;
  const float* hr = h + (size_t)t * C;
  float ssum = 0.f;
  for (int e = 0; e < E_; ++e) ssum += cap[e];
  float bids[E_];
  for (int e = 0; e < E_; ++e) {
    float dc = 0.f, da = 0.f;
    for (int c = 0; c < C; ++c) {
      float hv = hr[c];
      dc += hv * confw[(size_t)e * C + c];
      da += hv * cWa[(size_t)c * E_ + e];
    }
    float conf = 1.f / (1.f + __expf(-(dc + confb[e])));
    bids[e] = conf * (cap[e] / ssum) + da;
  }
  int i1 = 0; float v1 = bids[0];
  for (int e = 1; e < E_; ++e) if (bids[e] > v1) { v1 = bids[e]; i1 = e; }
  int i2 = -1; float v2 = -3.4e38f;
  for (int e = 0; e < E_; ++e) if (e != i1 && bids[e] > v2) { v2 = bids[e]; i2 = e; }
  float w1 = 1.f / (1.f + __expf(v2 - v1));
  for (int e = 0; e < E_; ++e) gate[(size_t)t * E_ + e] = 0.f;
  gate[(size_t)t * E_ + i1] = w1;
  gate[(size_t)t * E_ + i2] = 1.f - w1;
}

// ---------------- TDM tile loader ------------------------------------------
// Loads a rows x 32-element (u16) tile, row stride = stride_elems, from global
// into contiguous LDS at lds_dst.  D# bit layout per CDNA5 ISA 8.3/8.4:
//   group0: [1:0]=count=1, [63:32]=lds_addr, [120:64]=global_addr, [127:126]=2
//   group1: [17:16]=data_size(2B), dims/tile/stride fields as below.
// Tensor dims set generously (2^30 x 2^20): every tile is fully in range by
// construction (M % 128 == 0, K % 32 == 0, B rows padded to 64).
// This toolchain exposes the 6-arg builtin form:
//   (uint32x4 g0, int32x8 g1, int32x4 g2, int32x4 g3, int32x8 g4, i32 cpol)

#if HAVE_TDM
typedef int i32x8 __attribute__((ext_vector_type(8)));
typedef int i32x4 __attribute__((ext_vector_type(4)));

__device__ __forceinline__ void tdm_load_2d(const u16* gsrc, const u16* lds_dst,
                                            int stride_elems, int rows) {
  unsigned long long ga = (unsigned long long)(const void*)gsrc;
  v4u g0;
  g0.x = 1u;                                              // count=1 (valid, user)
  g0.y = (unsigned)(unsigned long long)(const void*)lds_dst; // lds_addr bytes
  g0.z = (unsigned)ga;                                    // global_addr[31:0]
  g0.w = (unsigned)((ga >> 32) & 0x01FFFFFFu) | (2u << 30); // addr[56:32]|type=2
  i32x8 g1;
  g1[0] = 0x00010000;               // wg_mask=0, data_size=1 (2 bytes)
  g1[1] = 0;                        // barrier addr=0, tensor_dim0 low16 (2^30)
  g1[2] = 0x4000;                   // tensor_dim0 hi16, tensor_dim1 low16 (2^20)
  g1[3] = 0x10 | (TILE_K << 16);    // tensor_dim1 hi16, tile_dim0 = 32
  g1[4] = rows;                     // tile_dim1 = rows, tile_dim2 = 0
  g1[5] = stride_elems;             // tensor_dim0_stride low32
  g1[6] = 0;                        // stride hi, dim1_stride low (unused, 2D)
  g1[7] = 0;
  i32x4 z4 = (i32x4)0;
  i32x8 z8 = (i32x8)0;
  __builtin_amdgcn_tensor_load_to_lds(g0, g1, z4, z4, z8, 0);
}
#endif

// ---------------- bf16 WMMA GEMM -------------------------------------------
// C[M,N] (+)= rowscale[m] * A[M,K] * Bt^T   (Bt stored transposed [Npad][K]).
// 128x64 tile / workgroup, 8 waves (4x2), each wave -> 32x32 via 4 WMMAs.
// Double-buffered LDS: wave 0 issues stage s+1's two TDM loads into the
// alternate buffer, then waits tensorcnt<=2 (stage s complete, s+1 in
// flight), so the DMA overlaps the WMMAs.  The two workgroup barriers per
// stage provide the producer/consumer hand-off between buffers (the buffer
// DMA'd at stage s was last read at stage s-1, before its bottom barrier).
// Requires: M % 128 == 0, K % 32 == 0, Bt rows padded to multiple of 64.

union FragBF { v4u q[2]; v16bf v; };

__global__ __launch_bounds__(256)
void k_gemm_bf16(const u16* __restrict__ A, const u16* __restrict__ Bt,
                 float* __restrict__ C, int M, int N, int K,
                 int accumulate, const float* __restrict__ row_scale, int scale_stride) {
  __shared__ alignas(16) u16 lA[2][TILE_M * TILE_K];   // row-major [m][k]
  __shared__ alignas(16) u16 lB[2][TILE_N * TILE_K];   // row-major [n][k]

  const int tid  = threadIdx.x;
  const int lane = tid & 31;
  const int wave = tid >> 5;
  const int wm = (wave >> 1) * 32;                  // 0,32,64,96
  const int wn = (wave & 1) * 32;                   // 0,32
  const int block_m = blockIdx.y * TILE_M;
  const int block_n = blockIdx.x * TILE_N;

  v8f acc00 = {}, acc01 = {}, acc10 = {}, acc11 = {};

  const u16* Ablk = A  + (size_t)block_m * K;       // M % TILE_M == 0 by contract
  const u16* Bblk = Bt + (size_t)block_n * K;       // rows padded -> in range

  const int nstage = K / TILE_K;

#if HAVE_TDM
  if (wave == 0) {                                  // prologue: stage 0 DMA
    tdm_load_2d(Ablk, lA[0], K, TILE_M);
    tdm_load_2d(Bblk, lB[0], K, TILE_N);
  }
#endif

  for (int s = 0; s < nstage; ++s) {
    const int buf = s & 1;
    const int k0 = s * TILE_K;
#if HAVE_TDM
    if (wave == 0) {
      if (s + 1 < nstage) {                         // pipeline next stage
        tdm_load_2d(Ablk + k0 + TILE_K, lA[buf ^ 1], K, TILE_M);
        tdm_load_2d(Bblk + k0 + TILE_K, lB[buf ^ 1], K, TILE_N);
        __builtin_amdgcn_s_wait_tensorcnt(2);       // stage s done, s+1 in flight
      } else {
        __builtin_amdgcn_s_wait_tensorcnt(0);       // drain
      }
    }
#else
    {
      int ra = tid >> 1, ka = (tid & 1) * 16;       // A: 128 rows x 32, 16/thr
      const u16* sp = Ablk + (size_t)ra * K + k0 + ka;
      *(v4u*)&lA[buf][ra * TILE_K + ka]     = *(const v4u*)(sp);
      *(v4u*)&lA[buf][ra * TILE_K + ka + 8] = *(const v4u*)(sp + 8);
      int rb = tid >> 2, kb = (tid & 3) * 8;        // B: 64 rows x 32, 8/thr
      *(v4u*)&lB[buf][rb * TILE_K + kb] =
          *(const v4u*)(Bblk + (size_t)rb * K + k0 + kb);
      if (s + 1 < nstage) {
        __builtin_prefetch(Ablk + (size_t)ra * K + k0 + TILE_K + ka, 0, 0);
        __builtin_prefetch(Bblk + (size_t)rb * K + k0 + TILE_K + kb, 0, 0);
      }
    }
#endif
    __syncthreads();

    // fragments per documented 16-bit A/B VGPR layouts
    const u16* la = lA[buf];
    const u16* lb = lB[buf];
    FragBF fa0, fa1, fb0, fb1;
    const int kh  = (lane >> 4) * 8;      // lanes 0-15: K 0..7/16..23; hi: 8../24..
    const int ar0 = (wm + (lane & 15)) * TILE_K;
    fa0.q[0] = *(const v4u*)&la[ar0 + kh];
    fa0.q[1] = *(const v4u*)&la[ar0 + 16 + kh];
    const int ar1 = ar0 + 16 * TILE_K;
    fa1.q[0] = *(const v4u*)&la[ar1 + kh];
    fa1.q[1] = *(const v4u*)&la[ar1 + 16 + kh];
    const int kb2 = (lane >> 4) * 8;      // lanes 0-15: K 0..7/16..23; hi: 8../24..
    const int br0 = (wn + (lane & 15)) * TILE_K;
    fb0.q[0] = *(const v4u*)&lb[br0 + kb2];
    fb0.q[1] = *(const v4u*)&lb[br0 + kb2 + 16];
    const int br1 = br0 + 16 * TILE_K;
    fb1.q[0] = *(const v4u*)&lb[br1 + kb2];
    fb1.q[1] = *(const v4u*)&lb[br1 + kb2 + 16];

    acc00 = __builtin_amdgcn_wmma_f32_16x16x32_bf16(false, fa0.v, false, fb0.v,
                                                    (short)0, acc00, false, false);
    acc01 = __builtin_amdgcn_wmma_f32_16x16x32_bf16(false, fa0.v, false, fb1.v,
                                                    (short)0, acc01, false, false);
    acc10 = __builtin_amdgcn_wmma_f32_16x16x32_bf16(false, fa1.v, false, fb0.v,
                                                    (short)0, acc10, false, false);
    acc11 = __builtin_amdgcn_wmma_f32_16x16x32_bf16(false, fa1.v, false, fb1.v,
                                                    (short)0, acc11, false, false);
    __syncthreads();
  }

  // epilogue: C/D layout -> lane n = lane&15, m = (lane>=16 ? 8:0) + vgpr idx
  const int n_loc = lane & 15;
  const int m_hi  = (lane >> 4) * 8;
  const int gn0 = block_n + wn + n_loc;
  const int gn1 = gn0 + 16;
#pragma unroll
  for (int r = 0; r < 8; ++r) {
    int gm0 = block_m + wm + m_hi + r;
    int gm1 = gm0 + 16;
    float s0 = row_scale ? row_scale[(size_t)gm0 * scale_stride] : 1.0f;
    float s1 = row_scale ? row_scale[(size_t)gm1 * scale_stride] : 1.0f;
    if (gn0 < N) {
      size_t o0 = (size_t)gm0 * N + gn0;
      float v0 = acc00[r] * s0; C[o0] = accumulate ? (C[o0] + v0) : v0;
      size_t o1 = (size_t)gm1 * N + gn0;
      float v1 = acc10[r] * s1; C[o1] = accumulate ? (C[o1] + v1) : v1;
    }
    if (gn1 < N) {
      size_t o0 = (size_t)gm0 * N + gn1;
      float v0 = acc01[r] * s0; C[o0] = accumulate ? (C[o0] + v0) : v0;
      size_t o1 = (size_t)gm1 * N + gn1;
      float v1 = acc11[r] * s1; C[o1] = accumulate ? (C[o1] + v1) : v1;
    }
  }
}

// ---------------------------------------------------------------------------

extern "C" void kernel_launch(void* const* d_in, const int* in_sizes, int n_in,
                              void* d_out, int out_size, void* d_ws, size_t ws_size,
                              hipStream_t stream) {
  (void)in_sizes; (void)n_in; (void)out_size; (void)ws_size;

  const int*   idx   = (const int*)  d_in[0];
  const float* cap   = (const float*)d_in[1];
  const float* embW  = (const float*)d_in[2];
  const float* headW = (const float*)d_in[3];
  const float* ln1s  = (const float*)d_in[4];
  const float* ln1b  = (const float*)d_in[5];
  const float* ln2s  = (const float*)d_in[6];
  const float* ln2b  = (const float*)d_in[7];
  const float* Wr    = (const float*)d_in[8];
  const float* Wk    = (const float*)d_in[9];
  const float* Wv    = (const float*)d_in[10];
  const float* Wo    = (const float*)d_in[11];
  const float* decay = (const float*)d_in[12];
  const float* W1    = (const float*)d_in[13];
  const float* W2    = (const float*)d_in[14];
  const float* confw = (const float*)d_in[15];
  const float* confb = (const float*)d_in[16];
  const float* trA   = (const float*)d_in[17];
  const float* trP   = (const float*)d_in[18];
  const float* trB   = (const float*)d_in[19];
  /* d_in[20] = critic_Wd : unused by reference output */
  const float* cWa   = (const float*)d_in[21];
  const float* brW   = (const float*)d_in[22];
  const float* lnos  = (const float*)d_in[23];
  const float* lnob  = (const float*)d_in[24];
  float* out = (float*)d_out;

  // ---- workspace bump allocator ----
  char* ws = (char*)d_ws;
  size_t off = 0;
  auto au16 = [&](size_t n) -> u16* {
    u16* p = (u16*)(ws + off); off += (n * sizeof(u16) + 255) & ~(size_t)255; return p;
  };
  auto af32 = [&](size_t n) -> float* {
    float* p = (float*)(ws + off); off += (n * sizeof(float) + 255) & ~(size_t)255; return p;
  };

  // bf16 transposed weights ([Npad][K] per matrix)
  u16* bWr = au16((size_t)L_ * C_ * C_);
  u16* bWk = au16((size_t)L_ * C_ * C_);
  u16* bWv = au16((size_t)L_ * C_ * C_);
  u16* bWo = au16((size_t)L_ * C_ * C_);
  u16* bW1 = au16((size_t)L_ * ER_ * H_ * C_);    // per expert: [H][C]
  u16* bW2 = au16((size_t)L_ * ER_ * C_ * H_);    // per expert: [C][H]
  u16* bA  = au16((size_t)L_ * ET_ * H2_ * C_);   // [H2][C]
  u16* bP  = au16((size_t)L_ * ET_ * H2_ * C_);   // [H2][C]
  u16* bB  = au16((size_t)L_ * ET_ * C_ * H2_);   // [C][H2]
  u16* bBr = au16((size_t)2 * C_ * C_);           // two [C][C] halves
  u16* bHd = au16((size_t)VP_ * C_);              // [VP][C], rows >= V zeroed

  // f32 activations
  float* x      = af32((size_t)BT_ * C_);
  float* hln    = af32((size_t)BT_ * C_);
  float* rbuf   = af32((size_t)BT_ * C_);
  float* kbuf   = af32((size_t)BT_ * C_);
  float* vbuf   = af32((size_t)BT_ * C_);
  float* vfirst = af32((size_t)BT_ * C_);
  float* states = af32((size_t)BT_ * C_);
  float* preff  = af32((size_t)BT_ * C_);
  float* tmpf   = af32((size_t)BT_ * H_);         // expert hidden; [a|pz] reuse
  float* gate   = af32((size_t)BT_ * E_);

  // bf16 activations
  u16* lnbf  = au16((size_t)BT_ * C_);
  u16* rsbf  = au16((size_t)BT_ * C_);
  u16* stbf  = au16((size_t)BT_ * C_);
  u16* prbf  = au16((size_t)BT_ * C_);
  u16* tmpbf = au16((size_t)BT_ * H_);

  auto grid1 = [](size_t n) {
    size_t b = (n + 255) / 256; if (b > 16384) b = 16384; return dim3((unsigned)b);
  };
  auto cvtT = [&](const float* s, u16* d, int batch, int K, int N, int Npad) {
    k_wT_bf16<<<grid1((size_t)batch * Npad * K), 256, 0, stream>>>(s, d, batch, K, N, Npad);
  };
  auto gemm = [&](const u16* A, const u16* Bt, float* Cp, int M, int N, int K,
                  int acc, const float* rs, int ss) {
    dim3 g((N + TILE_N - 1) / TILE_N, (M + TILE_M - 1) / TILE_M);
    k_gemm_bf16<<<g, 256, 0, stream>>>(A, Bt, Cp, M, N, K, acc, rs, ss);
  };

  // ---- weight conversion (transpose + bf16, once per launch) ----
  cvtT(Wr,   bWr, L_,        C_,  C_,  C_);
  cvtT(Wk,   bWk, L_,        C_,  C_,  C_);
  cvtT(Wv,   bWv, L_,        C_,  C_,  C_);
  cvtT(Wo,   bWo, L_,        C_,  C_,  C_);
  cvtT(W1,   bW1, L_ * ER_,  C_,  H_,  H_);
  cvtT(W2,   bW2, L_ * ER_,  H_,  C_,  C_);
  cvtT(trA,  bA,  L_ * ET_,  C_,  H2_, H2_);
  cvtT(trP,  bP,  L_ * ET_,  C_,  H2_, H2_);
  cvtT(trB,  bB,  L_ * ET_,  H2_, C_,  C_);
  cvtT(brW,  bBr, 2,         C_,  C_,  C_);
  cvtT(headW, bHd, 1,        C_,  V_,  VP_);

  const size_t nBTC = (size_t)BT_ * C_;

  // ---- embedding ----
  k_embed<<<grid1(nBTC), 256, 0, stream>>>(idx, embW, x, BT_, C_);

  for (int i = 0; i < L_; ++i) {
    const size_t wCC = (size_t)i * C_ * C_;

    // ln1 -> bf16 only
    k_layernorm<<<BT_, 256, 0, stream>>>(x, ln1s + (size_t)i * C_, ln1b + (size_t)i * C_,
                                         (float*)nullptr, lnbf, C_);
    // r,k,v projections
    gemm(lnbf, bWr + wCC, rbuf, BT_, C_, C_, 0, nullptr, 0);
    gemm(lnbf, bWk + wCC, kbuf, BT_, C_, C_, 0, nullptr, 0);
    gemm(lnbf, bWv + wCC, vbuf, BT_, C_, C_, 0, nullptr, 0);
    // v_first residual mixing
    k_vmix<<<grid1(nBTC), 256, 0, stream>>>(vbuf, vfirst, i, nBTC);
    // decayed kv recurrence
    k_scan<<<(B_ * C_ + 255) / 256, 256, 0, stream>>>(kbuf, vbuf, decay + (size_t)i * C_,
                                                      states, B_, T_, C_);
    // sigmoid(r)*states -> bf16, then @Wo accumulated into x (residual)
    k_rs_bf<<<grid1(nBTC), 256, 0, stream>>>(rbuf, states, rsbf, nBTC);
    gemm(rsbf, bWo + wCC, x, BT_, C_, C_, 1, nullptr, 0);

    // ln2 -> h (f32 for routing) + bf16 for GEMMs
    k_layernorm<<<BT_, 256, 0, stream>>>(x, ln2s + (size_t)i * C_, ln2b + (size_t)i * C_,
                                         hln, lnbf, C_);
    // routing -> dense gate [BT,E]
    k_route<<<(BT_ + 255) / 256, 256, 0, stream>>>(hln, confw + (size_t)i * E_ * C_,
                                                   confb + (size_t)i * E_,
                                                   cWa + (size_t)i * C_ * E_,
                                                   cap + (size_t)i * E_, gate, BT_, C_);
    // bridge prefix = tanh([h | state] @ bridge_W) via two GEMMs
    k_f2bf<<<grid1(nBTC), 256, 0, stream>>>(states, stbf, nBTC);
    gemm(lnbf, bBr,                   preff, BT_, C_, C_, 0, nullptr, 0);
    gemm(stbf, bBr + (size_t)C_ * C_, preff, BT_, C_, C_, 1, nullptr, 0);
    k_tanh_bf<<<grid1(nBTC), 256, 0, stream>>>(preff, prbf, nBTC);

    // ---- RWKV FFN experts: x += gate_e * ((relu(h@W1)^2) @ W2) ----
    for (int e = 0; e < ER_; ++e) {
      const u16* w1 = bW1 + ((size_t)i * ER_ + e) * H_ * C_;
      const u16* w2 = bW2 + ((size_t)i * ER_ + e) * C_ * H_;
      gemm(lnbf, w1, tmpf, BT_, H_, C_, 0, nullptr, 0);
      k_relu2_bf<<<grid1((size_t)BT_ * H_), 256, 0, stream>>>(tmpf, tmpbf, (size_t)BT_ * H_);
      gemm(tmpbf, w2, x, BT_, C_, H_, 1, gate + e, E_);
    }
    // ---- transformer experts: x += gate_{6+e} * ((silu(h@A)*(prefix@P)) @ B) ----
    for (int e = 0; e < ET_; ++e) {
      const u16* wa = bA + ((size_t)i * ET_ + e) * H2_ * C_;
      const u16* wp = bP + ((size_t)i * ET_ + e) * H2_ * C_;
      const u16* wb = bB + ((size_t)i * ET_ + e) * C_ * H2_;
      float* afb = tmpf;                       // [BT,H2]
      float* pfb = tmpf + (size_t)BT_ * H2_;   // [BT,H2]
      gemm(lnbf, wa, afb, BT_, H2_, C_, 0, nullptr, 0);
      gemm(prbf, wp, pfb, BT_, H2_, C_, 0, nullptr, 0);
      k_silu_mul_bf<<<grid1((size_t)BT_ * H2_), 256, 0, stream>>>(afb, pfb, tmpbf,
                                                                  (size_t)BT_ * H2_);
      gemm(tmpbf, wb, x, BT_, C_, H2_, 1, gate + (ER_ + e), E_);
    }
  }

  // ---- final LN + head projection [BT,C] @ [C,V] ----
  k_layernorm<<<BT_, 256, 0, stream>>>(x, lnos, lnob, (float*)nullptr, lnbf, C_);
  gemm(lnbf, bHd, out, BT_, V_, C_, 0, nullptr, 0);
}
